// QuantMixture_76794015252482
// MI455X (gfx1250) — 16-bit and 8-bit matrix operand order repaired and checked by reading, compiled, not run on hardware
//
#include <hip/hip_runtime.h>
#include <cstdint>
#include <cmath>

typedef int      v8i  __attribute__((ext_vector_type(8)));
typedef float    v8f  __attribute__((ext_vector_type(8)));
typedef _Float16 v16h __attribute__((ext_vector_type(16)));
typedef int      gv4i __attribute__((vector_size(16)));

#define BSZ   2
#define SEQ   2048
#define DM    2048
#define NH    16
#define NKV   8
#define HD    128
#define INTER 8192
#define TTOK  (BSZ*SEQ)
#define QKVD  (NH*HD + 2*NKV*HD)
#define EPSF  1e-6f
#define SCAP  50.0f
#define THETA 10000.0f

#define AS1 __attribute__((address_space(1)))
#define AS3 __attribute__((address_space(3)))
#if defined(__gfx1250__) && \
    __has_builtin(__builtin_amdgcn_global_load_async_to_lds_b128) && \
    __has_builtin(__builtin_amdgcn_s_wait_asynccnt)
#define USE_ASYNC_LDS 1
#define ASYNC_CP16(gsrc, ldst) \
    __builtin_amdgcn_global_load_async_to_lds_b128((AS1 gv4i*)(gsrc), \
                                                   (AS3 gv4i*)(ldst), 0, 0)
#else
#define USE_ASYNC_LDS 0
#endif

__device__ inline float fast_tanh(float x) {
    x = fminf(fmaxf(x, -15.f), 15.f);
    float e = __expf(2.0f * x);
    return (e - 1.0f) / (e + 1.0f);
}

template <bool IS_MAX>
__device__ inline float block_reduce(float v, float* sm) {
#pragma unroll
    for (int o = 16; o >= 1; o >>= 1) {
        float t = __shfl_xor(v, o, 32);
        v = IS_MAX ? fmaxf(v, t) : (v + t);
    }
    __syncthreads();
    int lane = threadIdx.x & 31, w = threadIdx.x >> 5;
    if (lane == 0) sm[w] = v;
    __syncthreads();
    if (threadIdx.x == 0) {
        float r = sm[0];
#pragma unroll
        for (int i = 1; i < 8; ++i) r = IS_MAX ? fmaxf(r, sm[i]) : (r + sm[i]);
        sm[0] = r;
    }
    __syncthreads();
    return sm[0];
}

__global__ void __launch_bounds__(256)
quant_rows_kernel(const float* __restrict__ w, int8_t* __restrict__ q,
                  float* __restrict__ s, int C) {
    __shared__ float sm[8];
    const int r = blockIdx.x;
    const float* row = w + (size_t)r * C;
    float mx = 0.f;
    for (int j = threadIdx.x; j < C; j += 256) mx = fmaxf(mx, fabsf(row[j]));
    mx = block_reduce<true>(mx, sm);
    float sc = fmaxf(mx / 127.0f, 1e-30f);
    if (threadIdx.x == 0) s[r] = sc;
    float rs = 1.0f / sc;
    for (int j = threadIdx.x; j < C; j += 256) {
        float qv = rintf(row[j] * rs);
        qv = fminf(fmaxf(qv, -127.f), 127.f);
        q[(size_t)r * C + j] = (int8_t)qv;
    }
}

__global__ void __launch_bounds__(256)
rmsnorm_quant_kernel(const float* __restrict__ x, const float* __restrict__ w,
                     int8_t* __restrict__ q, float* __restrict__ s, int C) {
    __shared__ float sm[8];
    const int t = blockIdx.x;
    const float* row = x + (size_t)t * C;
    const int per = C >> 8;
    float vals[8];
    float ss = 0.f;
#pragma unroll
    for (int j = 0; j < 8; ++j)
        if (j < per) { vals[j] = row[threadIdx.x + j * 256]; ss += vals[j] * vals[j]; }
    ss = block_reduce<false>(ss, sm);
    float inv = rsqrtf(ss / (float)C + EPSF);
    float mx = 0.f;
#pragma unroll
    for (int j = 0; j < 8; ++j)
        if (j < per) { vals[j] = vals[j] * inv * w[threadIdx.x + j * 256]; mx = fmaxf(mx, fabsf(vals[j])); }
    mx = block_reduce<true>(mx, sm);
    float sc = fmaxf(mx / 127.0f, 1e-8f);
    if (threadIdx.x == 0) s[t] = sc;
    float rs = 1.0f / sc;
#pragma unroll
    for (int j = 0; j < 8; ++j)
        if (j < per) {
            float qv = fminf(fmaxf(rintf(vals[j] * rs), -127.f), 127.f);
            q[(size_t)t * C + threadIdx.x + j * 256] = (int8_t)qv;
        }
}

__global__ void __launch_bounds__(256)
quant_token_kernel(const float* __restrict__ x, int8_t* __restrict__ q,
                   float* __restrict__ s, int C) {
    __shared__ float sm[8];
    const int t = blockIdx.x;
    const float* row = x + (size_t)t * C;
    const int per = C >> 8;
    float vals[8];
    float mx = 0.f;
#pragma unroll
    for (int j = 0; j < 8; ++j)
        if (j < per) { vals[j] = row[threadIdx.x + j * 256]; mx = fmaxf(mx, fabsf(vals[j])); }
    mx = block_reduce<true>(mx, sm);
    float sc = fmaxf(mx / 127.0f, 1e-8f);
    if (threadIdx.x == 0) s[t] = sc;
    float rs = 1.0f / sc;
#pragma unroll
    for (int j = 0; j < 8; ++j)
        if (j < per) {
            float qv = fminf(fmaxf(rintf(vals[j] * rs), -127.f), 127.f);
            q[(size_t)t * C + threadIdx.x + j * 256] = (int8_t)qv;
        }
}

__global__ void __launch_bounds__(256)
gelumul_quant_kernel(const float* __restrict__ gu, int8_t* __restrict__ q,
                     float* __restrict__ s) {
    __shared__ float sm[8];
    const int t = blockIdx.x;
    const float* gate = gu + (size_t)t * (2 * INTER);
    const float* up   = gate + INTER;
    float vals[32];
    float mx = 0.f;
#pragma unroll
    for (int j = 0; j < 32; ++j) {
        int idx = threadIdx.x + j * 256;
        float g = gate[idx];
        float inner = 0.7978845608028654f * (g + 0.044715f * g * g * g);
        float a = 0.5f * g * (1.0f + fast_tanh(inner)) * up[idx];
        vals[j] = a;
        mx = fmaxf(mx, fabsf(a));
    }
    mx = block_reduce<true>(mx, sm);
    float sc = fmaxf(mx / 127.0f, 1e-8f);
    if (threadIdx.x == 0) s[t] = sc;
    float rs = 1.0f / sc;
#pragma unroll
    for (int j = 0; j < 32; ++j) {
        float qv = fminf(fmaxf(rintf(vals[j] * rs), -127.f), 127.f);
        q[(size_t)t * INTER + threadIdx.x + j * 256] = (int8_t)qv;
    }
}

#define GBM 128
#define GBN 128
#define GBK 64
#define GLDA 80

__global__ void __launch_bounds__(256)
gemm_i8_kernel(const int8_t* __restrict__ A, const int8_t* __restrict__ B,
               const float* __restrict__ sa, const float* __restrict__ sb,
               const float* __restrict__ res, float* __restrict__ out,
               int M, int N, int K) {
#if USE_ASYNC_LDS
    __shared__ __align__(16) unsigned char lds[2][2][GBM * GLDA];
#else
    __shared__ __align__(16) unsigned char lds[1][2][GBM * GLDA];
#endif
    const int tid  = threadIdx.x;
    const int lane = tid & 31;
    const int wave = tid >> 5;
    const int wm   = wave >> 2;
    const int wn   = wave & 3;
    const int half = lane >> 4;
    const int mrow = lane & 15;
    const int tileM = blockIdx.y * GBM;
    const int tileN = blockIdx.x * GBN;

    v8i acc[4][2];
#pragma unroll
    for (int mi = 0; mi < 4; ++mi)
#pragma unroll
        for (int ni = 0; ni < 2; ++ni)
            acc[mi][ni] = (v8i){0, 0, 0, 0, 0, 0, 0, 0};

    const int lr = tid >> 2;
    const int cb = (tid & 3) * 16;
    const int8_t* gA = A + (size_t)(tileM + lr) * K + cb;
    const int8_t* gB = B + (size_t)(tileN + lr) * K + cb;

    auto compute_tile = [&](const unsigned char* tA, const unsigned char* tB) {
        v8i afr[4];
#pragma unroll
        for (int mi = 0; mi < 4; ++mi) {
            const unsigned char* base = tA + (wm * 64 + mi * 16 + mrow) * GLDA;
#pragma unroll
            for (int i = 0; i < 8; ++i) {
                int kk = ((i >> 1) & 1) * 16 + (i & 1) * 4 + half * 8 + (i >> 2) * 32;
                afr[mi][i] = *(const int*)(base + kk);
            }
        }
        v8i bfr[2];
#pragma unroll
        for (int ni = 0; ni < 2; ++ni) {
            const unsigned char* base = tB + (wn * 32 + ni * 16 + mrow) * GLDA;
#pragma unroll
            for (int i = 0; i < 8; ++i) {
                int kk = ((i >> 1) & 1) * 16 + (i & 1) * 4 + half * 8 + (i >> 2) * 32;
                bfr[ni][i] = *(const int*)(base + kk);
            }
        }
#pragma unroll
        for (int mi = 0; mi < 4; ++mi)
#pragma unroll
            for (int ni = 0; ni < 2; ++ni)
                acc[mi][ni] = __builtin_amdgcn_wmma_i32_16x16x64_iu8(
                    true, afr[mi], true, bfr[ni], acc[mi][ni], false, false);
    };

    const int ntiles = K / GBK;
#if USE_ASYNC_LDS
    {
        ASYNC_CP16(gA,                  &lds[0][0][lr * GLDA + cb]);
        ASYNC_CP16(gA + (size_t)64 * K, &lds[0][0][(lr + 64) * GLDA + cb]);
        ASYNC_CP16(gB,                  &lds[0][1][lr * GLDA + cb]);
        ASYNC_CP16(gB + (size_t)64 * K, &lds[0][1][(lr + 64) * GLDA + cb]);
    }
    for (int t = 0; t < ntiles; ++t) {
        const int cur = t & 1;
        if (t + 1 < ntiles) {
            const size_t k0 = (size_t)(t + 1) * GBK;
            ASYNC_CP16(gA + k0,                  &lds[cur ^ 1][0][lr * GLDA + cb]);
            ASYNC_CP16(gA + (size_t)64 * K + k0, &lds[cur ^ 1][0][(lr + 64) * GLDA + cb]);
            ASYNC_CP16(gB + k0,                  &lds[cur ^ 1][1][lr * GLDA + cb]);
            ASYNC_CP16(gB + (size_t)64 * K + k0, &lds[cur ^ 1][1][(lr + 64) * GLDA + cb]);
            __builtin_amdgcn_s_wait_asynccnt(4);
        } else {
            __builtin_amdgcn_s_wait_asynccnt(0);
        }
        __syncthreads();
        compute_tile(&lds[cur][0][0], &lds[cur][1][0]);
        __syncthreads();
    }
#else
    for (int t = 0; t < ntiles; ++t) {
        const size_t k0 = (size_t)t * GBK;
        __syncthreads();
#pragma unroll
        for (int rr = 0; rr < 2; ++rr) {
            uint4 va = *(const uint4*)(gA + (size_t)rr * 64 * K + k0);
            uint32_t* da = (uint32_t*)(&lds[0][0][(lr + rr * 64) * GLDA + cb]);
            da[0] = va.x; da[1] = va.y; da[2] = va.z; da[3] = va.w;
            uint4 vb = *(const uint4*)(gB + (size_t)rr * 64 * K + k0);
            uint32_t* db = (uint32_t*)(&lds[0][1][(lr + rr * 64) * GLDA + cb]);
            db[0] = vb.x; db[1] = vb.y; db[2] = vb.z; db[3] = vb.w;
        }
        __syncthreads();
        compute_tile(&lds[0][0][0], &lds[0][1][0]);
    }
#endif

#pragma unroll
    for (int mi = 0; mi < 4; ++mi)
#pragma unroll
        for (int ni = 0; ni < 2; ++ni) {
            const int n = tileN + wn * 32 + ni * 16 + mrow;
            const float sbn = sb[n];
            const int mbase = tileM + wm * 64 + mi * 16 + half * 8;
            const float* sap = sa + mbase;
            const size_t obase = (size_t)mbase * N + n;
            if (res != nullptr) {
#pragma unroll
                for (int r = 0; r < 8; ++r)
                    out[obase + (size_t)r * N] =
                        (float)acc[mi][ni][r] * sap[r] * sbn + res[obase + (size_t)r * N];
            } else {
#pragma unroll
                for (int r = 0; r < 8; ++r)
                    out[obase + (size_t)r * N] = (float)acc[mi][ni][r] * sap[r] * sbn;
            }
        }
}

__global__ void __launch_bounds__(256)
rope_pack_kernel(const float* __restrict__ qkv, _Float16* __restrict__ qh,
                 _Float16* __restrict__ kh, _Float16* __restrict__ vh) {
    unsigned idx = blockIdx.x * 256u + threadIdx.x;
    int d2 = idx & 63;  idx >>= 6;
    int slot = idx & 31; idx >>= 5;
    int s = idx & (SEQ - 1);
    int b = idx >> 11;
    const size_t trow = (size_t)(b * SEQ + s) * QKVD;
    float ex = (float)(2 * d2) / (float)HD;
    float invf = powf(THETA, -ex);
    float f = (float)s * invf;
    float c = cosf(f), sn = sinf(f);
    if (slot < 16) {
        const float* src = qkv + trow + slot * HD;
        float x0 = src[d2], x1 = src[d2 + 64];
        _Float16* dst = qh + ((size_t)(b * NH + slot) * SEQ + s) * HD;
        dst[d2]      = (_Float16)(x0 * c - x1 * sn);
        dst[d2 + 64] = (_Float16)(x1 * c + x0 * sn);
    } else if (slot < 24) {
        int h = slot - 16;
        const float* src = qkv + trow + NH * HD + h * HD;
        float x0 = src[d2], x1 = src[d2 + 64];
        _Float16* dst = kh + ((size_t)(b * NKV + h) * SEQ + s) * HD;
        dst[d2]      = (_Float16)(x0 * c - x1 * sn);
        dst[d2 + 64] = (_Float16)(x1 * c + x0 * sn);
    } else {
        int h = slot - 24;
        const float* src = qkv + trow + NH * HD + NKV * HD + h * HD;
        _Float16* dst = vh + ((size_t)(b * NKV + h) * SEQ + s) * HD;
        dst[d2]      = (_Float16)src[d2];
        dst[d2 + 64] = (_Float16)src[d2 + 64];
    }
}

#define FQT 64
#define FKT 32
#define KST 132
#define VST 36

__global__ void __launch_bounds__(128)
flash_attn_kernel(const _Float16* __restrict__ qh, const _Float16* __restrict__ kh,
                  const _Float16* __restrict__ vh, float* __restrict__ out) {
    __shared__ _Float16 lK[FKT * KST];
    __shared__ _Float16 lVT[HD * VST];
    __shared__ _Float16 lP[4][16 * FKT];

    const int q0   = blockIdx.x * FQT;
    const int bh   = blockIdx.y;
    const int b    = bh / NH;
    const int h    = bh % NH;
    const int kvh  = h >> 1;
    const int tid  = threadIdx.x;
    const int lane = tid & 31;
    const int wave = tid >> 5;
    const int half = lane >> 4;
    const int mrow = lane & 15;

    const _Float16* qbase = qh + ((size_t)(b * NH + h) * SEQ + (q0 + wave * 16 + mrow)) * HD;
    const _Float16* kbase = kh + (size_t)(b * NKV + kvh) * SEQ * HD;
    const _Float16* vbase = vh + (size_t)(b * NKV + kvh) * SEQ * HD;

    v16h qfr[4];
#pragma unroll
    for (int ks = 0; ks < 4; ++ks)
#pragma unroll
        for (int i = 0; i < 8; ++i) {
            int kp = (i < 4) ? (half * 8 + 2 * i) : (16 + half * 8 + 2 * (i - 4));
            int k = ks * 32 + kp;
            qfr[ks][2 * i]     = qbase[k];
            qfr[ks][2 * i + 1] = qbase[k + 1];
        }

    v8f ao[8];
#pragma unroll
    for (int nd = 0; nd < 8; ++nd) ao[nd] = (v8f){0, 0, 0, 0, 0, 0, 0, 0};
    float m_st[8], l_st[8];
#pragma unroll
    for (int r = 0; r < 8; ++r) { m_st[r] = -INFINITY; l_st[r] = 0.f; }

    const float rscale = 0.08838834764831845f;
    const int kend = q0 + FQT;

    for (int kb = 0; kb < kend; kb += FKT) {
        __syncthreads();
        for (int idx = tid; idx < FKT * 64; idx += 128) {
            int row = idx >> 6, dw = idx & 63;
            size_t src = ((size_t)(kb + row)) * HD + dw * 2;
            uint32_t wk = *(const uint32_t*)(kbase + src);
            *(uint32_t*)(&lK[row * KST + dw * 2]) = wk;
            union { uint32_t u; _Float16 hh[2]; } cv;
            cv.u = *(const uint32_t*)(vbase + src);
            lVT[(dw * 2) * VST + row]     = cv.hh[0];
            lVT[(dw * 2 + 1) * VST + row] = cv.hh[1];
        }
        __syncthreads();

        v8f sf[2];
#pragma unroll
        for (int ni = 0; ni < 2; ++ni) {
            sf[ni] = (v8f){0, 0, 0, 0, 0, 0, 0, 0};
            const _Float16* kr = &lK[(ni * 16 + mrow) * KST];
#pragma unroll
            for (int ks = 0; ks < 4; ++ks) {
                v16h bfr;
#pragma unroll
                for (int i = 0; i < 8; ++i) {
                    int k = ks * 32 + ((i < 4) ? (half * 8 + 2 * i) : (16 + half * 8 + 2 * (i - 4)));
                    bfr[2 * i]     = kr[k];
                    bfr[2 * i + 1] = kr[k + 1];
                }
                sf[ni] = __builtin_amdgcn_wmma_f32_16x16x32_f16(
                    false, qfr[ks], false, bfr, (short)0, sf[ni], false, false);
            }
        }

#pragma unroll
        for (int ni = 0; ni < 2; ++ni)
#pragma unroll
            for (int r = 0; r < 8; ++r) {
                float xv = sf[ni][r] * rscale;
                xv = fast_tanh(xv * (1.0f / SCAP)) * SCAP;
                int qrow = q0 + wave * 16 + r + half * 8;
                int krow = kb + ni * 16 + mrow;
                if (krow > qrow) xv = -1e9f;
                sf[ni][r] = xv;
            }

#pragma unroll
        for (int r = 0; r < 8; ++r) {
            float mx = fmaxf(sf[0][r], sf[1][r]);
#pragma unroll
            for (int o = 8; o >= 1; o >>= 1) mx = fmaxf(mx, __shfl_xor(mx, o, 32));
            float mnew = fmaxf(m_st[r], mx);
            float alpha = __expf(m_st[r] - mnew);
            float rowsum = 0.f;
#pragma unroll
            for (int ni = 0; ni < 2; ++ni) {
                float p = __expf(sf[ni][r] - mnew);
                sf[ni][r] = p;
                rowsum += p;
            }
#pragma unroll
            for (int o = 8; o >= 1; o >>= 1) rowsum += __shfl_xor(rowsum, o, 32);
            l_st[r] = l_st[r] * alpha + rowsum;
            m_st[r] = mnew;
#pragma unroll
            for (int nd = 0; nd < 8; ++nd) ao[nd][r] *= alpha;
        }

#pragma unroll
        for (int ni = 0; ni < 2; ++ni)
#pragma unroll
            for (int r = 0; r < 8; ++r)
                lP[wave][(r + half * 8) * FKT + ni * 16 + mrow] = (_Float16)sf[ni][r];
        __syncthreads();

        v16h pfr;
#pragma unroll
        for (int i = 0; i < 8; ++i) {
            int kp = (i < 4) ? (half * 8 + 2 * i) : (16 + half * 8 + 2 * (i - 4));
            pfr[2 * i]     = lP[wave][mrow * FKT + kp];
            pfr[2 * i + 1] = lP[wave][mrow * FKT + kp + 1];
        }
#pragma unroll
        for (int nd = 0; nd < 8; ++nd) {
            const _Float16* vr = &lVT[(nd * 16 + mrow) * VST];
            v16h vfr;
#pragma unroll
            for (int i = 0; i < 8; ++i) {
                int k = (i < 4) ? (half * 8 + 2 * i) : (16 + half * 8 + 2 * (i - 4));
                vfr[2 * i]     = vr[k];
                vfr[2 * i + 1] = vr[k + 1];
            }
            ao[nd] = __builtin_amdgcn_wmma_f32_16x16x32_f16(
                false, pfr, false, vfr, (short)0, ao[nd], false, false);
        }
    }

#pragma unroll
    for (int nd = 0; nd < 8; ++nd)
#pragma unroll
        for (int r = 0; r < 8; ++r) {
            int m = r + half * 8;
            int srow = q0 + wave * 16 + m;
            float v = ao[nd][r] / l_st[r];
            out[(size_t)(b * SEQ + srow) * (NH * HD) + h * HD + nd * 16 + mrow] = v;
        }
}

extern "C" void kernel_launch(void* const* d_in, const int* in_sizes, int n_in,
                              void* d_out, int out_size, void* d_ws, size_t ws_size,
                              hipStream_t stream) {
    (void)in_sizes; (void)n_in; (void)out_size; (void)ws_size;
    const float* hidden = (const float*)d_in[0];
    const float* w_in_n = (const float*)d_in[1];
    const float* w_po_n = (const float*)d_in[2];
    const float* wqkv   = (const float*)d_in[3];
    const float* wo     = (const float*)d_in[4];
    const float* wgu    = (const float*)d_in[5];
    const float* wdn    = (const float*)d_in[6];
    float* out = (float*)d_out;

    char* ws = (char*)d_ws;
    size_t off = 0;
    auto alloc = [&](size_t bytes) -> char* {
        char* p = ws + off;
        off = (off + bytes + 255) & ~(size_t)255;
        return p;
    };
    int8_t* q_wqkv = (int8_t*)alloc((size_t)QKVD * DM);
    int8_t* q_wo   = (int8_t*)alloc((size_t)DM * (NH * HD));
    int8_t* q_wgu  = (int8_t*)alloc((size_t)2 * INTER * DM);
    int8_t* q_wdn  = (int8_t*)alloc((size_t)DM * INTER);
    float*  s_wqkv = (float*)alloc(sizeof(float) * QKVD);
    float*  s_wo   = (float*)alloc(sizeof(float) * DM);
    float*  s_wgu  = (float*)alloc(sizeof(float) * 2 * INTER);
    float*  s_wdn  = (float*)alloc(sizeof(float) * DM);
    int8_t* qact   = (int8_t*)alloc((size_t)TTOK * INTER);
    float*  sact   = (float*)alloc(sizeof(float) * TTOK);
    float*  xbuf   = (float*)alloc(sizeof(float) * (size_t)TTOK * DM);
    float*  attn   = (float*)alloc(sizeof(float) * (size_t)TTOK * NH * HD);
    _Float16* qhb  = (_Float16*)alloc(sizeof(_Float16) * (size_t)BSZ * NH * SEQ * HD);
    _Float16* khb  = (_Float16*)alloc(sizeof(_Float16) * (size_t)BSZ * NKV * SEQ * HD);
    _Float16* vhb  = (_Float16*)alloc(sizeof(_Float16) * (size_t)BSZ * NKV * SEQ * HD);
    float*  big    = (float*)alloc(sizeof(float) * (size_t)TTOK * 2 * INTER);
    float*  qkvbuf = big;

    quant_rows_kernel<<<QKVD, 256, 0, stream>>>(wqkv, q_wqkv, s_wqkv, DM);
    quant_rows_kernel<<<DM, 256, 0, stream>>>(wo, q_wo, s_wo, NH * HD);
    quant_rows_kernel<<<2 * INTER, 256, 0, stream>>>(wgu, q_wgu, s_wgu, DM);
    quant_rows_kernel<<<DM, 256, 0, stream>>>(wdn, q_wdn, s_wdn, INTER);

    rmsnorm_quant_kernel<<<TTOK, 256, 0, stream>>>(hidden, w_in_n, qact, sact, DM);
    gemm_i8_kernel<<<dim3(QKVD / GBN, TTOK / GBM), 256, 0, stream>>>(
        qact, q_wqkv, sact, s_wqkv, nullptr, qkvbuf, TTOK, QKVD, DM);
    rope_pack_kernel<<<(BSZ * SEQ * 32 * 64) / 256, 256, 0, stream>>>(qkvbuf, qhb, khb, vhb);
    flash_attn_kernel<<<dim3(SEQ / FQT, BSZ * NH), 128, 0, stream>>>(qhb, khb, vhb, attn);
    quant_token_kernel<<<TTOK, 256, 0, stream>>>(attn, qact, sact, NH * HD);
    gemm_i8_kernel<<<dim3(DM / GBN, TTOK / GBM), 256, 0, stream>>>(
        qact, q_wo, sact, s_wo, hidden, xbuf, TTOK, DM, NH * HD);

    rmsnorm_quant_kernel<<<TTOK, 256, 0, stream>>>(xbuf, w_po_n, qact, sact, DM);
    gemm_i8_kernel<<<dim3(2 * INTER / GBN, TTOK / GBM), 256, 0, stream>>>(
        qact, q_wgu, sact, s_wgu, nullptr, big, TTOK, 2 * INTER, DM);
    gelumul_quant_kernel<<<TTOK, 256, 0, stream>>>(big, qact, sact);
    gemm_i8_kernel<<<dim3(DM / GBN, TTOK / GBM), 256, 0, stream>>>(
        qact, q_wdn, sact, s_wdn, xbuf, out, TTOK, DM, INTER);
}
